// MMD_loss_33449205301987
// MI455X (gfx1250) — matrix operand order repaired and checked by reading, not compiled, not run on hardware
//
#include <hip/hip_runtime.h>
#include <cmath>

typedef __attribute__((ext_vector_type(16))) __bf16       v16bf;
typedef __attribute__((ext_vector_type(8)))  float        v8f;
typedef __attribute__((ext_vector_type(8)))  unsigned int v8u;

#define NROWS 8192
#define DIM   128
#define BLK_M 256                      // rows per workgroup
#define BLK_N 128                      // cols per workgroup
#define GRID_X (NROWS / BLK_M)         // 32
#define GRID_Y (NROWS / BLK_N)         // 64
#define BLOCKS_PER_Z (GRID_X * GRID_Y) // 2048
#define PAIRS_PER_MAT (NROWS * DIM / 2)

// ---------------- bf16 split helpers ----------------
__device__ __forceinline__ unsigned bf16_rne(float f) {
    unsigned u = __float_as_uint(f);
    return (u + 0x7FFFu + ((u >> 16) & 1u)) >> 16;
}

// Two consecutive-K f32 -> packed bf16 pair (low K in low 16 bits): hi + residual lo.
__device__ __forceinline__ void cvt_pair(float a, float b, unsigned& hi, unsigned& lo) {
    unsigned h0 = bf16_rne(a), h1 = bf16_rne(b);
    float r0 = a - __uint_as_float(h0 << 16);
    float r1 = b - __uint_as_float(h1 << 16);
    hi = h0 | (h1 << 16);
    lo = bf16_rne(r0) | (bf16_rne(r1) << 16);
}

__device__ __forceinline__ v16bf as_v16bf(v8u x) {
    union { v8u u; v16bf b; } c; c.u = x; return c.b;
}

// A-matrix 16x32 bf16 frag (ISA 7.12.2): lane m=l&15, h=l>>4.
// dwords j=0..3 hold K=8h..8h+7, j=4..7 hold K=16+8h..16+8h+7 -> two b128 loads.
__device__ __forceinline__ v8u load_fragA(const unsigned* __restrict__ xp,
                                          int row0, int kc, int lane) {
    int m = lane & 15, h = lane >> 4;
    const unsigned* row = xp + ((size_t)(row0 + m) * DIM + kc) / 2;  // dword units
    uint4 a = *(const uint4*)(row + 4 * h);        // K = 8h..8h+7
    uint4 b = *(const uint4*)(row + 8 + 4 * h);    // K = 16+8h..16+8h+7
    v8u f;
    f[0] = a.x; f[1] = a.y; f[2] = a.z; f[3] = a.w;
    f[4] = b.x; f[5] = b.y; f[6] = b.z; f[7] = b.w;
    return f;
}

// B-matrix 32x16 bf16 frag: lane n=l&15, h=l>>4; dwords j=0..3 hold K=8h..8h+7,
// j=4..7 hold K=16+8h..16+8h+7 (same as A) -> two b128 loads.
__device__ __forceinline__ v8u load_fragB(const unsigned* __restrict__ xp,
                                          int col0, int kc, int lane) {
    int n = lane & 15, h = lane >> 4;
    const unsigned* row = xp + ((size_t)(col0 + n) * DIM + kc) / 2;  // dword units
    uint4 a = *(const uint4*)(row + 4 * h);        // K = 8h..8h+7
    uint4 b = *(const uint4*)(row + 8 + 4 * h);    // K = 16+8h..16+8h+7
    v8u f;
    f[0] = a.x; f[1] = a.y; f[2] = a.z; f[3] = a.w;
    f[4] = b.x; f[5] = b.y; f[6] = b.z; f[7] = b.w;
    return f;
}

// ---------------- prep 1: f32 -> packed bf16 hi/lo arrays ----------------
__global__ void mmd_convert_kernel(const float* __restrict__ x1,
                                   const float* __restrict__ x2,
                                   unsigned* __restrict__ h1, unsigned* __restrict__ l1,
                                   unsigned* __restrict__ h2, unsigned* __restrict__ l2) {
    int i = blockIdx.x * blockDim.x + threadIdx.x;      // pair index
    if (i >= 2 * PAIRS_PER_MAT) return;
    const float* x; unsigned *hd, *ld; int p;
    if (i < PAIRS_PER_MAT) { x = x1; hd = h1; ld = l1; p = i; }
    else                   { x = x2; hd = h2; ld = l2; p = i - PAIRS_PER_MAT; }
    float2 v = *(const float2*)(x + (size_t)2 * p);
    unsigned hi, lo; cvt_pair(v.x, v.y, hi, lo);
    hd[p] = hi; ld[p] = lo;
}

// ---------------- prep 2: row squared norms (from original f32) ----------------
__global__ void mmd_prep_kernel(const float* __restrict__ x1,
                                const float* __restrict__ x2,
                                float* __restrict__ n1,
                                float* __restrict__ n2) {
    int i = blockIdx.x * blockDim.x + threadIdx.x;
    if (i >= 2 * NROWS) return;
    const float* x = (i < NROWS) ? x1 : x2;
    float* o       = (i < NROWS) ? n1 : n2;
    int r = i & (NROWS - 1);
    const float4* row = (const float4*)(x + (size_t)r * DIM);
    float s = 0.f;
#pragma unroll 8
    for (int j = 0; j < DIM / 4; ++j) {
        float4 v = row[j];
        s = fmaf(v.x, v.x, fmaf(v.y, v.y, fmaf(v.z, v.z, fmaf(v.w, v.w, s))));
    }
    o[r] = s;
}

// ---------------- main: tiled WMMA kernel-mean ----------------
__global__ void __launch_bounds__(256)
mmd_tile_kernel(const unsigned* __restrict__ h1, const unsigned* __restrict__ l1,
                const unsigned* __restrict__ h2, const unsigned* __restrict__ l2,
                const float* __restrict__ n1, const float* __restrict__ n2,
                double* __restrict__ parts, float gamma) {
    int z = blockIdx.z;  // 0:(x1,x1)=m11  1:(x2,x2)=m22  2:(x1,x2)=m12
    const unsigned* ah = (z == 1) ? h2 : h1;
    const unsigned* al = (z == 1) ? l2 : l1;
    const unsigned* bh = (z == 0) ? h1 : h2;
    const unsigned* bl = (z == 0) ? l1 : l2;
    const float* na = (z == 1) ? n2 : n1;
    const float* nb = (z == 0) ? n1 : n2;

    int lane = threadIdx.x & 31;
    int wave = threadIdx.x >> 5;            // 0..7
    int wm = wave & 3;                      // 4 waves along M
    int wn = wave >> 2;                     // 2 waves along N
    int base_m = blockIdx.x * BLK_M + wm * 64;
    int base_n = blockIdx.y * BLK_N + wn * 64;

    v8f acc[4][4];
#pragma unroll
    for (int tm = 0; tm < 4; ++tm)
#pragma unroll
        for (int tn = 0; tn < 4; ++tn) acc[tm][tn] = (v8f){};

    for (int kc = 0; kc < DIM; kc += 32) {
        v8u Ah[4], Al[4], Bh[4], Bl[4];
#pragma unroll
        for (int tm = 0; tm < 4; ++tm) {
            Ah[tm] = load_fragA(ah, base_m + 16 * tm, kc, lane);
            Al[tm] = load_fragA(al, base_m + 16 * tm, kc, lane);
        }
#pragma unroll
        for (int tn = 0; tn < 4; ++tn) {
            Bh[tn] = load_fragB(bh, base_n + 16 * tn, kc, lane);
            Bl[tn] = load_fragB(bl, base_n + 16 * tn, kc, lane);
        }
#pragma unroll
        for (int tm = 0; tm < 4; ++tm) {
            v16bf vah = as_v16bf(Ah[tm]), val = as_v16bf(Al[tm]);
#pragma unroll
            for (int tn = 0; tn < 4; ++tn) {
                v16bf vbh = as_v16bf(Bh[tn]), vbl = as_v16bf(Bl[tn]);
                // a·b ≈ ah·bh + ah·bl + al·bh   (lo·lo ~2^-18 rel., dropped)
                acc[tm][tn] = __builtin_amdgcn_wmma_f32_16x16x32_bf16(
                    false, vah, false, vbh, (short)0, acc[tm][tn], false, false);
                acc[tm][tn] = __builtin_amdgcn_wmma_f32_16x16x32_bf16(
                    false, vah, false, vbl, (short)0, acc[tm][tn], false, false);
                acc[tm][tn] = __builtin_amdgcn_wmma_f32_16x16x32_bf16(
                    false, val, false, vbh, (short)0, acc[tm][tn], false, false);
            }
        }
    }

    // Epilogue in log2 domain:
    //   exp(-g*max(sq,0)) = exp2(min(g2*acc + ca[r] + cb[tn], 0))
    //   with ca = -g*log2e*||a||^2, cb = -g*log2e*||b||^2, g2 = 2*g*log2e.
    // C/D layout: lane l, VGPR r -> M = r + 8*(l>>4), N = l&15.
    const float LOG2E = 1.4426950408889634f;
    float gl = gamma * LOG2E;
    float g2 = 2.0f * gl;
    int h = lane >> 4, nn = lane & 15;

    float cb[4];
#pragma unroll
    for (int tn = 0; tn < 4; ++tn) cb[tn] = -gl * nb[base_n + 16 * tn + nn];

    float p0 = 0.f, p1 = 0.f, p2 = 0.f, p3 = 0.f;  // fixed partition -> deterministic
#pragma unroll
    for (int tm = 0; tm < 4; ++tm) {
        float ca[8];
#pragma unroll
        for (int r = 0; r < 8; ++r) ca[r] = -gl * na[base_m + 16 * tm + 8 * h + r];
#pragma unroll
        for (int tn = 0; tn < 4; ++tn) {
            float psum = 0.f;
#pragma unroll
            for (int r = 0; r < 8; ++r) {
                float t = fmaf(g2, acc[tm][tn][r], ca[r] + cb[tn]);
                t = fminf(t, 0.0f);                     // == max(sq,0) clamp
                psum += __builtin_amdgcn_exp2f(t);      // raw v_exp_f32
            }
            if (tn == 0) p0 += psum;
            else if (tn == 1) p1 += psum;
            else if (tn == 2) p2 += psum;
            else p3 += psum;
        }
    }
    float partial = (p0 + p1) + (p2 + p3);

    // wave32 reduction, then deterministic per-block partial (no FP atomics)
#pragma unroll
    for (int off = 16; off > 0; off >>= 1)
        partial += __shfl_down(partial, off, 32);

    __shared__ double bsum[8];
    if (lane == 0) bsum[wave] = (double)partial;
    __syncthreads();
    if (threadIdx.x == 0) {
        double s = 0.0;
        for (int w = 0; w < 8; ++w) s += bsum[w];
        parts[(size_t)z * BLOCKS_PER_Z + blockIdx.y * GRID_X + blockIdx.x] = s;
    }
}

// ---------------- final deterministic reduction ----------------
__global__ void mmd_final_kernel(const double* __restrict__ parts,
                                 float* __restrict__ out) {
    __shared__ double sh[192];
    int t = threadIdx.x;        // 0..191
    int z = t >> 6;             // 0..2
    int c = t & 63;             // 0..63
    double s = 0.0;
    const int CH = BLOCKS_PER_Z / 64; // 32
    for (int i = 0; i < CH; ++i)
        s += parts[(size_t)z * BLOCKS_PER_Z + c * CH + i];
    sh[t] = s;
    __syncthreads();
    if (t == 0) {
        double m[3] = {0.0, 0.0, 0.0};
        for (int zz = 0; zz < 3; ++zz)
            for (int k = 0; k < 64; ++k) m[zz] += sh[zz * 64 + k];
        double inv = 1.0 / ((double)NROWS * (double)NROWS);
        double v = (m[0] - 2.0 * m[2] + m[1]) * inv;  // m11 - 2*m12 + m22
        out[0] = (float)sqrt(v > 0.0 ? v : 0.0);
    }
}

// ---------------- host launcher ----------------
extern "C" void kernel_launch(void* const* d_in, const int* in_sizes, int n_in,
                              void* d_out, int out_size, void* d_ws, size_t ws_size,
                              hipStream_t stream) {
    const float* x1 = (const float*)d_in[0];
    const float* x2 = (const float*)d_in[1];
    float* out = (float*)d_out;

    // ws layout: h1|l1|h2|l2 (each NROWS*DIM/2 dwords = 2MB) | n1,n2 f32 | parts f64
    char* ws = (char*)d_ws;
    const size_t matBytes = (size_t)PAIRS_PER_MAT * sizeof(unsigned); // 2MB
    unsigned* h1 = (unsigned*)(ws);
    unsigned* l1 = (unsigned*)(ws + matBytes);
    unsigned* h2 = (unsigned*)(ws + 2 * matBytes);
    unsigned* l2 = (unsigned*)(ws + 3 * matBytes);
    float*    n1 = (float*)(ws + 4 * matBytes);
    float*    n2 = (float*)(ws + 4 * matBytes + (size_t)NROWS * sizeof(float));
    double* parts = (double*)(ws + 4 * matBytes + (size_t)2 * NROWS * sizeof(float));

    // gamma = 1/(2*gz^2), gz = 2*Gamma((d+1)/2)/Gamma(d/2), d = 128 (both inputs)
    double d = (double)DIM;
    double gz = 2.0 * std::exp(std::lgamma(0.5 * (d + 1.0)) - std::lgamma(0.5 * d));
    float gamma = (float)(1.0 / (2.0 * gz * gz));

    mmd_convert_kernel<<<(2 * PAIRS_PER_MAT + 255) / 256, 256, 0, stream>>>(
        x1, x2, h1, l1, h2, l2);
    mmd_prep_kernel<<<(2 * NROWS + 255) / 256, 256, 0, stream>>>(x1, x2, n1, n2);

    dim3 grid(GRID_X, GRID_Y, 3);
    mmd_tile_kernel<<<grid, 256, 0, stream>>>(h1, l1, h2, l2, n1, n2, parts, gamma);

    mmd_final_kernel<<<1, 192, 0, stream>>>(parts, out);
}
